// STAttentionBlock_63136019251511
// MI455X (gfx1250) — matrix operand order repaired and checked by reading, not compiled, not run on hardware
//
#include <hip/hip_runtime.h>
#include <hip/hip_bf16.h>

// CDNA5 / gfx1250, wave32.  Matrix path: v_wmma_f32_16x16x32_bf16.
// Async path: global_load_async_to_lds_b128 + s_wait_asynccnt (ASYNCcnt).
typedef __attribute__((ext_vector_type(16))) __bf16 v16bf;
typedef __attribute__((ext_vector_type(8)))  float  v8f;

#define P_SP 1600   // T*V = 64*25 positions per (n, channel)

__device__ __forceinline__ __bf16 f2bf(float f) { return (__bf16)f; }

// Async global->LDS copy of 16 bytes (4 floats).  VDST VGPR carries the LDS
// byte offset (low 32 bits of the generic shared pointer), VADDR the global
// address.  Tracked with ASYNCcnt.
__device__ __forceinline__ void async_copy16(const float* g, float* l) {
  unsigned lds_off = (unsigned)(uintptr_t)l;
  asm volatile("global_load_async_to_lds_b128 %0, %1, off"
               :: "v"(lds_off), "v"(g) : "memory");
}
__device__ __forceinline__ void async_wait0() {
  asm volatile("s_wait_asynccnt 0x0" ::: "memory");
}

// ---------------------------------------------------------------------------
// Generic 1x1-conv GEMM:  Y[n,o,p] = epi( sum_c W[o,c] * X[n,c,p] )
// epi: v = acc*s + sh (+resid) (lrelu); s/sh fold bias + batchnorm.
// Block: 128 threads = 4 waves.  Block tile: 64 o x 64 p.
// Wave w owns o-subtile w (16 rows) and all 4 p-subtiles: per K-step one
// A fragment feeds 4 WMMAs (amortizes W loads 4x; B tile shared by 4 waves).
// ---------------------------------------------------------------------------
__global__ __launch_bounds__(128) void gemm_conv1x1(
    const float* __restrict__ X, const float* __restrict__ W,
    const float* __restrict__ bias, const float* __restrict__ gamma,
    const float* __restrict__ beta, const float* __restrict__ resid,
    float* __restrict__ Y, int Cin, int O, int do_lrelu)
{
  __shared__ __align__(16) __bf16 Bt[64][32];   // [position][k] bf16 (4 KB)
  const int n     = blockIdx.z;
  const int p0    = blockIdx.x * 64;
  const int tid   = threadIdx.x;
  const int wave  = tid >> 5;
  const int lane  = tid & 31;
  const int otile = blockIdx.y * 64 + wave * 16;
  const int col   = lane & 15;               // M for A-frag, N for B/C frags
  const int h     = lane >> 4;

  v8f acc0 = {}, acc1 = {}, acc2 = {}, acc3 = {};
  const int    m     = otile + col;
  const float* Wrow  = W + (size_t)m * Cin;
  const size_t xbase = ((size_t)n * Cin) * P_SP + p0;
  const int    spp   = tid & 63;             // staging: position
  const int    skh   = tid >> 6;             // staging: k parity (0..1)

  for (int k0 = 0; k0 < Cin; k0 += 32) {
    __syncthreads();
    // cooperative stage of B tile (64 p x 32 k); fixed 16-iteration unroll,
    // consecutive tids -> consecutive positions (coalesced rows)
    #pragma unroll
    for (int i = 0; i < 16; ++i) {
      int k = i * 2 + skh;
      Bt[spp][k] = f2bf(X[xbase + (size_t)(k0 + k) * P_SP + spp]);
    }
    __syncthreads();
    // A fragment (16x32 bf16): lane(m=col,h): K = 16h..16h+15 contiguous
    v16bf a;
    #pragma unroll
    for (int e = 0; e < 8; ++e) a[e]     = f2bf(Wrow[k0 + 16*h + e]);
    #pragma unroll
    for (int e = 0; e < 8; ++e) a[8 + e] = f2bf(Wrow[k0 + 16*h + 8 + e]);
    // 4 B fragments (32x16 bf16): lane(n=col,h): K = 16h..16h+15 contiguous
    v16bf b0, b1, b2, b3;
    #pragma unroll
    for (int e = 0; e < 16; ++e) {
      b0[e] = Bt[ 0 + col][16*h + e];
      b1[e] = Bt[16 + col][16*h + e];
      b2[e] = Bt[32 + col][16*h + e];
      b3[e] = Bt[48 + col][16*h + e];
    }
    acc0 = __builtin_amdgcn_wmma_f32_16x16x32_bf16(false, a, false, b0, (short)0, acc0, false, false);
    acc1 = __builtin_amdgcn_wmma_f32_16x16x32_bf16(false, a, false, b1, (short)0, acc1, false, false);
    acc2 = __builtin_amdgcn_wmma_f32_16x16x32_bf16(false, a, false, b2, (short)0, acc2, false, false);
    acc3 = __builtin_amdgcn_wmma_f32_16x16x32_bf16(false, a, false, b3, (short)0, acc3, false, false);
  }

  const float rs = rsqrtf(1.0f + 1e-5f);
  #pragma unroll
  for (int r = 0; r < 8; ++r) {
    int o = otile + r + 8*h;                 // C/D layout: VGPR r -> row r+8h
    float s, sh;
    if (gamma) { s = gamma[o] * rs; sh = bias[o] * s + beta[o]; }
    else       { s = 1.0f;          sh = bias[o]; }
    const size_t yb = ((size_t)n * O + o) * P_SP + p0 + col;
    float vr[4] = { acc0[r], acc1[r], acc2[r], acc3[r] };
    #pragma unroll
    for (int j = 0; j < 4; ++j) {
      float v = vr[j] * s + sh;
      size_t yi = yb + j * 16;
      if (resid)    v += resid[yi];
      if (do_lrelu) v = v > 0.0f ? v : 0.1f * v;
      Y[yi] = v;
    }
  }
}

// ---------------------------------------------------------------------------
// Spatial attention: att[n,s,u,v] = tanh(sum_{c,t} q.k / 4096)*alpha + att0
// q = qk[n, s*64+c], k = qk[n, 192+s*64+c].  One block per (s, n).
// LDS tiles filled with async load-to-LDS (1600-float contiguous slabs).
// ---------------------------------------------------------------------------
__global__ __launch_bounds__(256) void spatial_att_k(
    const float* __restrict__ qk, const float* __restrict__ alphas,
    const float* __restrict__ att0s, float* __restrict__ att)
{
  __shared__ __align__(16) float qc[64][25];
  __shared__ __align__(16) float kc[64][25];
  const int s = blockIdx.x, n = blockIdx.y, tid = threadIdx.x;
  const size_t qb = ((size_t)n*384 + s*64) * P_SP;
  const size_t kb = ((size_t)n*384 + 192 + s*64) * P_SP;
  float a0 = 0.f, a1 = 0.f, a2 = 0.f;
  const int o0 = tid, o1 = tid + 256, o2r = tid + 512;
  const int o2 = (o2r < 625) ? o2r : 0;
  const int u0 = o0/25, v0 = o0%25;
  const int u1 = o1/25, v1 = o1%25;
  const int u2 = o2/25, v2 = o2%25;
  for (int f0 = 0; f0 < 4096; f0 += 64) {   // flat (c,t) index, c-major
    __syncthreads();
    // 2 x 1600 floats = 800 async b128 chunks
    for (int idx = tid; idx < 800; idx += 256) {
      if (idx < 400) async_copy16(qk + qb + (size_t)f0*25 + idx*4, &qc[0][0] + idx*4);
      else { int j = idx - 400;
             async_copy16(qk + kb + (size_t)f0*25 + j*4, &kc[0][0] + j*4); }
    }
    async_wait0();
    __syncthreads();
    for (int kk = 0; kk < 64; ++kk) {
      a0 += qc[kk][u0] * kc[kk][v0];
      a1 += qc[kk][u1] * kc[kk][v1];
      a2 += qc[kk][u2] * kc[kk][v2];
    }
  }
  const float alpha = alphas[s];
  const size_t ab = ((size_t)n*3 + s) * 625;
  att[ab + o0] = tanhf(a0 * (1.0f/4096.0f)) * alpha + att0s[s*625 + o0];
  if (o1  < 625) att[ab + o1]  = tanhf(a1 * (1.0f/4096.0f)) * alpha + att0s[s*625 + o1];
  if (o2r < 625) att[ab + o2r] = tanhf(a2 * (1.0f/4096.0f)) * alpha + att0s[s*625 + o2r];
}

// ---------------------------------------------------------------------------
// y[n, s*128+c, t, v] = sum_u x[n,c,t,u] * att[n,s,u,v].  Block per (s,n).
// ---------------------------------------------------------------------------
__global__ __launch_bounds__(256) void apply_spatial_k(
    const float* __restrict__ x, const float* __restrict__ att,
    float* __restrict__ y)
{
  __shared__ float al[25][25];
  const int s = blockIdx.x, n = blockIdx.y, tid = threadIdx.x;
  for (int idx = tid; idx < 625; idx += 256)
    al[idx/25][idx%25] = att[((size_t)n*3 + s)*625 + idx];
  __syncthreads();
  const size_t xb = (size_t)n * 128 * P_SP;
  const size_t yb = ((size_t)n*384 + s*128) * P_SP;
  for (int pos = tid; pos < 8192; pos += 256) {   // pos = c*64 + t
    float xv[25];
    #pragma unroll
    for (int u = 0; u < 25; ++u) xv[u] = x[xb + (size_t)pos*25 + u];
    for (int v = 0; v < 25; ++v) {
      float acc = 0.f;
      #pragma unroll
      for (int u = 0; u < 25; ++u) acc += xv[u] * al[u][v];
      y[yb + (size_t)pos*25 + v] = acc;
    }
  }
}

// ---------------------------------------------------------------------------
// Temporal attention: att_dir[n,s,t,q] = mask * alpha * tanh(q.k / 1600)
// dir 0 = forward (ch 0.. / 256.., mask t>=q), dir 1 = backward
// (ch 128.. / 384.., mask q>=t).  Block per (s, dir, n).
// ---------------------------------------------------------------------------
__global__ __launch_bounds__(256) void temporal_att_k(
    const float* __restrict__ qk2, const float* __restrict__ alphat_f,
    const float* __restrict__ alphat_b, float* __restrict__ attF,
    float* __restrict__ attB)
{
  __shared__ __align__(16) float qc[4][64][25];
  __shared__ __align__(16) float kc[4][64][25];
  const int s = blockIdx.x, dir = blockIdx.y, n = blockIdx.z, tid = threadIdx.x;
  const int qch = (dir == 0 ?   0 : 128) + s*64;
  const int kch = (dir == 0 ? 256 : 384) + s*64;
  const size_t qb = ((size_t)n*512 + qch) * P_SP;
  const size_t kb = ((size_t)n*512 + kch) * P_SP;
  float acc[16];
  #pragma unroll
  for (int i = 0; i < 16; ++i) acc[i] = 0.f;
  for (int c0 = 0; c0 < 64; c0 += 4) {
    __syncthreads();
    // 2 x 4 x 1600 floats = 3200 async b128 chunks
    for (int idx = tid; idx < 3200; idx += 256) {
      int which = idx / 1600;              // 0 = q, 1 = k
      int rem   = idx % 1600;
      int cc    = rem / 400;
      int cof   = rem % 400;
      const float* g = (which ? (qk2 + kb) : (qk2 + qb))
                       + (size_t)(c0 + cc) * P_SP + cof * 4;
      float* l = (which ? &kc[0][0][0] : &qc[0][0][0]) + cc * 1600 + cof * 4;
      async_copy16(g, l);
    }
    async_wait0();
    __syncthreads();
    #pragma unroll
    for (int i = 0; i < 16; ++i) {
      int o = tid + 256*i;
      int t = o >> 6, q = o & 63;
      float a = 0.f;
      for (int cc = 0; cc < 4; ++cc) {
        #pragma unroll
        for (int v = 0; v < 25; ++v) a += qc[cc][t][v] * kc[cc][q][v];
      }
      acc[i] += a;
    }
  }
  const float alpha = (dir == 0) ? alphat_f[s] : alphat_b[s];
  float* out = (dir == 0) ? attF : attB;
  #pragma unroll
  for (int i = 0; i < 16; ++i) {
    int o = tid + 256*i;
    int t = o >> 6, q = o & 63;
    bool keep = (dir == 0) ? (t >= q) : (q >= t);
    out[((size_t)n*2 + s)*4096 + o] =
        keep ? tanhf(acc[i] * (1.0f/1600.0f)) * alpha : 0.f;
  }
}

// ---------------------------------------------------------------------------
// z[n, dir*256 + s*128 + c, q, v] = sum_t y2[n,c,t,v] * att_dir[n,s,t,q]
// Block per (s, dir, n); att + y2 channel slab staged via async-to-LDS.
// ---------------------------------------------------------------------------
__global__ __launch_bounds__(256) void apply_temporal_k(
    const float* __restrict__ y2, const float* __restrict__ attF,
    const float* __restrict__ attB, float* __restrict__ z)
{
  __shared__ __align__(16) float al[64][64];
  __shared__ __align__(16) float yl[64][25];
  const int s = blockIdx.x, dir = blockIdx.y, n = blockIdx.z, tid = threadIdx.x;
  const float* att = ((dir == 0) ? attF : attB) + ((size_t)n*2 + s)*4096;
  // 4096 floats = 1024 async b128 chunks
  for (int idx = tid; idx < 1024; idx += 256)
    async_copy16(att + idx*4, &al[0][0] + idx*4);
  async_wait0();
  const size_t zb = ((size_t)n*512 + dir*256 + s*128) * P_SP;
  const size_t yb = (size_t)n * 128 * P_SP;
  for (int c = 0; c < 128; ++c) {
    __syncthreads();
    if (tid < 400) async_copy16(y2 + yb + (size_t)c*P_SP + tid*4, &yl[0][0] + tid*4);
    async_wait0();
    __syncthreads();
    for (int idx = tid; idx < 1600; idx += 256) {
      int q = idx / 25, v = idx % 25;
      float a = 0.f;
      #pragma unroll
      for (int t = 0; t < 64; ++t) a += yl[t][v] * al[t][q];
      z[zb + (size_t)c*P_SP + idx] = a;
    }
  }
}

// ---------------------------------------------------------------------------
// TCN 7x1 conv along T (+ bn + residual + lrelu), WMMA over (dt, c) K-loop.
// Shift by dt in flattened (t,v) space is +/- (dt-3)*25 whole positions.
// Same 64o x 64p block tile: 4 WMMAs per A fragment.
// ---------------------------------------------------------------------------
__global__ __launch_bounds__(128) void tcn_k(
    const float* __restrict__ Z, const float* __restrict__ Wt,
    const float* __restrict__ bias, const float* __restrict__ gamma,
    const float* __restrict__ beta, float* __restrict__ out)
{
  __shared__ __align__(16) __bf16 Bt[64][32];
  const int n     = blockIdx.z;
  const int p0    = blockIdx.x * 64;
  const int tid   = threadIdx.x;
  const int wave  = tid >> 5;
  const int lane  = tid & 31;
  const int otile = blockIdx.y * 64 + wave * 16;
  const int col   = lane & 15;
  const int h     = lane >> 4;
  const int m     = otile + col;
  const int spp   = tid & 63;
  const int skh   = tid >> 6;
  v8f acc0 = {}, acc1 = {}, acc2 = {}, acc3 = {};
  const size_t zbase = (size_t)n * 128 * P_SP;

  for (int dt = 0; dt < 7; ++dt) {
    const int shift = (dt - 3) * 25;
    const int  p    = p0 + spp + shift;
    const bool ok   = (p >= 0) && (p < P_SP);
    for (int k0 = 0; k0 < 128; k0 += 32) {
      __syncthreads();
      #pragma unroll
      for (int i = 0; i < 16; ++i) {
        int k = i * 2 + skh;
        Bt[spp][k] = f2bf(ok ? Z[zbase + (size_t)(k0+k)*P_SP + p] : 0.f);
      }
      __syncthreads();
      v16bf a;   // W_tcn layout (o, c, 7, 1): stride 7 along c
      #pragma unroll
      for (int e = 0; e < 8; ++e)
        a[e]     = f2bf(Wt[((size_t)m*128 + (k0 + 16*h + e))*7 + dt]);
      #pragma unroll
      for (int e = 0; e < 8; ++e)
        a[8 + e] = f2bf(Wt[((size_t)m*128 + (k0 + 16*h + 8 + e))*7 + dt]);
      v16bf b0, b1, b2, b3;
      #pragma unroll
      for (int e = 0; e < 16; ++e) {
        b0[e] = Bt[ 0 + col][16*h + e];
        b1[e] = Bt[16 + col][16*h + e];
        b2[e] = Bt[32 + col][16*h + e];
        b3[e] = Bt[48 + col][16*h + e];
      }
      acc0 = __builtin_amdgcn_wmma_f32_16x16x32_bf16(false, a, false, b0, (short)0, acc0, false, false);
      acc1 = __builtin_amdgcn_wmma_f32_16x16x32_bf16(false, a, false, b1, (short)0, acc1, false, false);
      acc2 = __builtin_amdgcn_wmma_f32_16x16x32_bf16(false, a, false, b2, (short)0, acc2, false, false);
      acc3 = __builtin_amdgcn_wmma_f32_16x16x32_bf16(false, a, false, b3, (short)0, acc3, false, false);
    }
  }
  const float rs = rsqrtf(1.0f + 1e-5f);
  #pragma unroll
  for (int r = 0; r < 8; ++r) {
    int o = otile + r + 8*h;
    float s  = gamma[o] * rs;
    float sh = bias[o] * s + beta[o];
    const size_t ob = ((size_t)n*128 + o) * P_SP + p0 + col;
    float vr[4] = { acc0[r], acc1[r], acc2[r], acc3[r] };
    #pragma unroll
    for (int j = 0; j < 4; ++j) {
      size_t idx = ob + j * 16;
      float v = vr[j] * s + sh + Z[idx];
      out[idx] = v > 0.0f ? v : 0.1f * v;
    }
  }
}

// ---------------------------------------------------------------------------
extern "C" void kernel_launch(void* const* d_in, const int* in_sizes, int n_in,
                              void* d_out, int out_size, void* d_ws, size_t ws_size,
                              hipStream_t stream)
{
  const float* x       = (const float*)d_in[0];
  const float* W_ins   = (const float*)d_in[1];
  const float* b_ins   = (const float*)d_in[2];
  const float* alphas  = (const float*)d_in[3];
  const float* att0s   = (const float*)d_in[4];
  const float* W_outs  = (const float*)d_in[5];
  const float* b_outs  = (const float*)d_in[6];
  const float* g_outs  = (const float*)d_in[7];
  const float* be_outs = (const float*)d_in[8];
  const float* W_ffs   = (const float*)d_in[9];
  const float* b_ffs   = (const float*)d_in[10];
  const float* g_ffs   = (const float*)d_in[11];
  const float* be_ffs  = (const float*)d_in[12];
  const float* W_int   = (const float*)d_in[13];
  const float* b_int   = (const float*)d_in[14];
  const float* alphat_b= (const float*)d_in[15];
  const float* alphat_f= (const float*)d_in[16];
  const float* W_outt  = (const float*)d_in[17];
  const float* b_outt  = (const float*)d_in[18];
  const float* g_outt  = (const float*)d_in[19];
  const float* be_outt = (const float*)d_in[20];
  const float* W_fft   = (const float*)d_in[21];
  const float* b_fft   = (const float*)d_in[22];
  const float* g_fft   = (const float*)d_in[23];
  const float* be_fft  = (const float*)d_in[24];
  const float* W_tcn   = (const float*)d_in[25];
  const float* b_tcn   = (const float*)d_in[26];
  const float* g_tcn   = (const float*)d_in[27];
  const float* be_tcn  = (const float*)d_in[28];

  // Workspace regions (bytes):
  //   rA: 209,715,200  (qk1 -> y -> qk2 -> z, reused sequentially)
  //   rB:  52,428,800  (y1 -> z1)     rC: 52,428,800 (y2, long-lived)
  //   rD:  52,428,800  (z2)           small: att_s / att_f / att_b
  char*  ws   = (char*)d_ws;
  float* rA   = (float*)(ws);
  float* rB   = (float*)(ws + 209715200ull);
  float* rC   = (float*)(ws + 209715200ull +     52428800ull);
  float* rD   = (float*)(ws + 209715200ull + 2ull*52428800ull);
  float* attS = (float*)(ws + 209715200ull + 3ull*52428800ull);
  float* attF = (float*)((char*)attS + 480000ull);
  float* attB = (float*)((char*)attF + 2097152ull);

  const dim3 blk(128);
  // 1) qk1 = W_ins @ x + b_ins                     (O=384, Cin=128) -> rA
  gemm_conv1x1<<<dim3(25,6,64), blk, 0, stream>>>(
      x, W_ins, b_ins, nullptr, nullptr, nullptr, rA, 128, 384, 0);
  // 2) spatial attention matrices (tanh/alpha/att0) -> attS
  spatial_att_k<<<dim3(3,64), dim3(256), 0, stream>>>(rA, alphas, att0s, attS);
  // 3) y = x @ att                                  -> rA (qk1 dead)
  apply_spatial_k<<<dim3(3,64), dim3(256), 0, stream>>>(x, attS, rA);
  // 4) y1 = lrelu(x + bn(W_outs @ y + b))          (O=128, Cin=384) -> rB
  gemm_conv1x1<<<dim3(25,2,64), blk, 0, stream>>>(
      rA, W_outs, b_outs, g_outs, be_outs, x, rB, 384, 128, 1);
  // 5) y2 = lrelu(x + bn(W_ffs @ y1 + b))          (128,128) -> rC
  gemm_conv1x1<<<dim3(25,2,64), blk, 0, stream>>>(
      rB, W_ffs, b_ffs, g_ffs, be_ffs, x, rC, 128, 128, 1);
  // 6) qk2 = W_int @ y2 + b                        (O=512) -> rA (y dead)
  gemm_conv1x1<<<dim3(25,8,64), blk, 0, stream>>>(
      rC, W_int, b_int, nullptr, nullptr, nullptr, rA, 128, 512, 0);
  // 7) temporal attention (fwd + bwd, masked tanh) -> attF/attB
  temporal_att_k<<<dim3(2,2,64), dim3(256), 0, stream>>>(
      rA, alphat_f, alphat_b, attF, attB);
  // 8) z = y2 applied through att_f/att_b          -> rA (qk2 dead)
  apply_temporal_k<<<dim3(2,2,64), dim3(256), 0, stream>>>(rC, attF, attB, rA);
  // 9) z1 = lrelu(y2 + bn(W_outt @ z + b))         (128,512) -> rB
  gemm_conv1x1<<<dim3(25,2,64), blk, 0, stream>>>(
      rA, W_outt, b_outt, g_outt, be_outt, rC, rB, 512, 128, 1);
  // 10) z2 = lrelu(y2 + bn(W_fft @ z1 + b))        (128,128) -> rD
  gemm_conv1x1<<<dim3(25,2,64), blk, 0, stream>>>(
      rB, W_fft, b_fft, g_fft, be_fft, rC, rD, 128, 128, 1);
  // 11) out = lrelu(z2 + bn(tcn(z2) + b))          -> d_out
  tcn_k<<<dim3(25,2,64), blk, 0, stream>>>(
      rD, W_tcn, b_tcn, g_tcn, be_tcn, (float*)d_out);
}
